// UCBAttention_51599737094421
// MI455X (gfx1250) — matrix operand order repaired and checked by reading, not compiled, not run on hardware
//
#include <hip/hip_runtime.h>
#include <hip/hip_bf16.h>
#include <math.h>

// Problem constants (fixed by the reference)
#define BATCH   4
#define NSEQ    1024
#define CDIM    768
#define NHEAD   12
#define DH      64
#define TOPK    128
#define QKVDIM  2304

typedef __attribute__((ext_vector_type(16))) _Float16 v16h;
typedef __attribute__((ext_vector_type(8)))  _Float16 v8h;
typedef __attribute__((ext_vector_type(8)))  float    v8f;

// ---------------------------------------------------------------------------
// WMMA helpers (CDNA5 gfx1250, wave32). D = A(16x32 f16) * B(32x16 f16) + C(f32)
// ---------------------------------------------------------------------------
__device__ __forceinline__ v8f wmma_f16(v16h a, v16h b, v8f c) {
    return __builtin_amdgcn_wmma_f32_16x16x32_f16(
        /*neg_a=*/false, a, /*neg_b=*/false, b,
        /*c_mod=*/(short)0, c, /*reuse_a=*/false, /*reuse_b=*/false);
}

// A fragment: 16x32 (MxK) f16, row-major source with leading dim ld.
// ISA layout: lane<16 holds M=lane, K={0..7}+{16..23}; lane>=16 holds K={8..15}+{24..31}.
// -> two contiguous 8-half (16B) chunks per lane.
__device__ __forceinline__ v16h load_a_frag(const _Float16* __restrict__ base,
                                            int ld, int k0, int lane) {
    int m = lane & 15, g = lane >> 4;
    const _Float16* p = base + (size_t)m * ld + k0 + g * 8;
    v8h lo = *(const v8h*)p;
    v8h hi = *(const v8h*)(p + 16);
    v16h r;
#pragma unroll
    for (int i = 0; i < 8; ++i) { r[i] = lo[i]; r[i + 8] = hi[i]; }
    return r;
}

// A fragment sourced from an f32 row in LDS (attention weights).
__device__ __forceinline__ v16h load_a_frag_lds(const float* base, int ld,
                                                int k0, int lane) {
    int m = lane & 15, g = lane >> 4;
    const float* p = base + m * ld + k0 + g * 8;
    v16h r;
#pragma unroll
    for (int i = 0; i < 8; ++i) {
        r[i]     = (_Float16)p[i];
        r[i + 8] = (_Float16)p[i + 16];
    }
    return r;
}

// B fragment: 32x16 (KxN) f16 where B[k][n] = mat[n][k] (mat row-major, ld).
// ISA layout: lane<16 -> N=lane, K={0..7}+{16..23}; lane>=16 -> N=lane-16, K={8..15}+{24..31}.
// -> two contiguous 8-half (16B) chunks per lane.
__device__ __forceinline__ v16h load_b_frag(const _Float16* __restrict__ base,
                                            int ld, int k0, int lane) {
    int n = lane & 15, g = lane >> 4;
    const _Float16* p = base + (size_t)n * ld + k0 + g * 8;
    v8h lo = *(const v8h*)p;
    v8h hi = *(const v8h*)(p + 16);
    v16h r;
#pragma unroll
    for (int i = 0; i < 8; ++i) { r[i] = lo[i]; r[i + 8] = hi[i]; }
    return r;
}

// ---------------------------------------------------------------------------
// wave32 reductions
// ---------------------------------------------------------------------------
__device__ __forceinline__ float wave_max(float v) {
#pragma unroll
    for (int o = 16; o; o >>= 1) v = fmaxf(v, __shfl_xor(v, o, 32));
    return v;
}
__device__ __forceinline__ float wave_min(float v) {
#pragma unroll
    for (int o = 16; o; o >>= 1) v = fminf(v, __shfl_xor(v, o, 32));
    return v;
}
__device__ __forceinline__ float wave_sum(float v) {
#pragma unroll
    for (int o = 16; o; o >>= 1) v += __shfl_xor(v, o, 32);
    return v;
}
__device__ __forceinline__ int wave_sumi(int v) {
#pragma unroll
    for (int o = 16; o; o >>= 1) v += __shfl_xor(v, o, 32);
    return v;
}

// ---------------------------------------------------------------------------
// Kernel 0: f32 -> f16 conversion (x, qkv_w, proj_w)
// ---------------------------------------------------------------------------
__global__ void ucb_cvt_f16_kernel(const float* __restrict__ src,
                                   _Float16* __restrict__ dst, int n) {
    for (int i = blockIdx.x * blockDim.x + threadIdx.x; i < n;
         i += gridDim.x * blockDim.x)
        dst[i] = (_Float16)src[i];
}

// ---------------------------------------------------------------------------
// Kernel 1: QKV GEMM. [4096 x 768] @ [2304 x 768]^T -> q/k (scaled q) + v^T
// One wave per 16x16 output tile; 8 waves per block.
// ---------------------------------------------------------------------------
__global__ void ucb_qkv_gemm_kernel(const _Float16* __restrict__ x_h,
                                    const _Float16* __restrict__ w_h,
                                    _Float16* __restrict__ q_h,
                                    _Float16* __restrict__ k_h,
                                    _Float16* __restrict__ v_t) {
    int lane = threadIdx.x & 31, wid = threadIdx.x >> 5;
    int ntile = blockIdx.x * 8 + wid;   // 0..143  (2304/16)
    int mtile = blockIdx.y;             // 0..255  (4096/16)
    const _Float16* abase = x_h + (size_t)mtile * 16 * CDIM;
    const _Float16* bbase = w_h + (size_t)ntile * 16 * CDIM;
    v8f acc = {};
    for (int k0 = 0; k0 < CDIM; k0 += 32) {
        v16h a = load_a_frag(abase, CDIM, k0, lane);
        v16h b = load_b_frag(bbase, CDIM, k0, lane);
        acc = wmma_f16(a, b, acc);
    }
    int n = lane & 15, g = lane >> 4;
    int col = ntile * 16 + n;
    int s = col / CDIM;                 // 0=q, 1=k, 2=v
    int rem = col - s * CDIM;
    int h = rem >> 6, d = rem & 63;
#pragma unroll
    for (int i = 0; i < 8; ++i) {
        int row = mtile * 16 + g * 8 + i;
        int b_ = row >> 10, ns = row & 1023;
        float val = acc[i];
        if (s == 0) {
            q_h[((size_t)(b_ * NHEAD + h) * NSEQ + ns) * DH + d] =
                (_Float16)(val * 0.125f);          // fold Dh^-0.5
        } else if (s == 1) {
            k_h[((size_t)(b_ * NHEAD + h) * NSEQ + ns) * DH + d] = (_Float16)val;
        } else {
            v_t[((size_t)(b_ * NHEAD + h) * DH + d) * NSEQ + ns] = (_Float16)val;
        }
    }
}

// ---------------------------------------------------------------------------
// Kernel 2: fused scores -> softmax -> UCB top-k -> renorm -> P@V + score_delta.
// Block = (h, 16-query tile); batch loop inside so counts are read once and
// score_delta written once. 256 threads = 8 waves. Dynamic LDS:
//   sc[16][1024] f32 (scores, then pruned weights)  64KB
//   expl[16][1024] f32 (UCB bonus, batch-invariant) 64KB
//   cnt[16][1024] u8  (mask sum over batch)         16KB
// ---------------------------------------------------------------------------
__global__ void ucb_attn_kernel(const _Float16* __restrict__ q_h,
                                const _Float16* __restrict__ k_h,
                                const _Float16* __restrict__ v_t,
                                const float* __restrict__ ucb_count,
                                const int* __restrict__ counter_p,
                                const int* __restrict__ enabled_p,
                                _Float16* __restrict__ ctx_h,
                                float* __restrict__ score_delta) {
    extern __shared__ unsigned char smem[];
    float* sc = (float*)smem;                          // 16*1024 f32
    float* expl = (float*)(smem + 16 * 1024 * 4);      // 16*1024 f32
    unsigned char* cnt = smem + 2 * 16 * 1024 * 4;     // 16*1024 u8

    int tid = threadIdx.x, lane = tid & 31, wid = tid >> 5;
    int qt = blockIdx.x;                // 0..63
    int h = blockIdx.y;                 // 0..11
    int counter = counter_p[0];
    bool use_ucb = (enabled_p[0] != 0) && (counter > 50);
    float logc = logf((float)counter + 1.0f);

    // expl (shared across batch) + delta counters
    for (int idx = tid; idx < 16 * 1024; idx += 256) {
        int r = idx >> 10, kk = idx & 1023;
        float e = 0.0f;
        if (use_ucb) {
            float c = ucb_count[((size_t)h * NSEQ + qt * 16 + r) * NSEQ + kk];
            e = sqrtf(logc / (c + 1e-6f));             // BETA = 1
        }
        expl[idx] = e;
        cnt[idx] = 0;
    }
    __syncthreads();

    for (int b = 0; b < BATCH; ++b) {
        size_t bh = (size_t)(b * NHEAD + h);
        const _Float16* qb = q_h + (bh * NSEQ + (size_t)qt * 16) * DH;

        // ---- QK^T scores: each wave covers 8 of 64 key tiles ----
        for (int kt = wid; kt < 64; kt += 8) {
            const _Float16* kb = k_h + (bh * NSEQ + (size_t)kt * 16) * DH;
            v8f acc = {};
#pragma unroll
            for (int k0 = 0; k0 < DH; k0 += 32) {
                v16h a = load_a_frag(qb, DH, k0, lane);
                v16h bm = load_b_frag(kb, DH, k0, lane);
                acc = wmma_f16(a, bm, acc);
            }
            int n = lane & 15, g = lane >> 4;
#pragma unroll
            for (int i = 0; i < 8; ++i)
                sc[(g * 8 + i) * NSEQ + kt * 16 + n] = acc[i];
        }
        __syncthreads();

        // ---- softmax + top-k bisection + renorm: 2 rows per wave ----
        for (int rr = 0; rr < 2; ++rr) {
            int r = wid * 2 + rr;
            float* srow = sc + r * NSEQ;
            const float* erow = expl + r * NSEQ;
            float u[32];
            float msc = -3.4e38f, lo = 3.4e38f, hi = -3.4e38f;
#pragma unroll
            for (int j = 0; j < 32; ++j) {
                float s = srow[lane + 32 * j];
                float uu = s + erow[lane + 32 * j];
                u[j] = uu;
                msc = fmaxf(msc, s);
                lo = fminf(lo, uu);
                hi = fmaxf(hi, uu);
            }
            msc = wave_max(msc);
            lo = wave_min(lo);
            hi = wave_max(hi);

            float thr = -3.4e38f;
            if (use_ucb) {
                // bisect for the 128th-largest UCB score
                for (int it = 0; it < 28; ++it) {
                    float mid = 0.5f * (lo + hi);
                    int c = 0;
#pragma unroll
                    for (int j = 0; j < 32; ++j) c += (u[j] >= mid) ? 1 : 0;
                    c = wave_sumi(c);
                    if (c >= TOPK) lo = mid; else hi = mid;
                }
                thr = lo;
            }

            float Z = 0.0f, Em = 0.0f;
#pragma unroll
            for (int j = 0; j < 32; ++j) {
                float e = __expf(srow[lane + 32 * j] - msc);
                Z += e;
                if (u[j] >= thr) Em += e;
            }
            Z = wave_sum(Z);
            Em = wave_sum(Em);
            // pruned/(pruned.sum+1e-8) == e*mask / (Em + 1e-8*Z)
            float inv = 1.0f / (Em + 1e-8f * Z);
#pragma unroll
            for (int j = 0; j < 32; ++j) {
                int kk = lane + 32 * j;
                bool m = (u[j] >= thr);
                float e = __expf(srow[kk] - msc);
                srow[kk] = m ? e * inv : 0.0f;
                if (m) cnt[r * NSEQ + kk] += 1;   // row owned by this wave
            }
        }
        __syncthreads();

        // ---- context = W @ V : waves 0..3 each own a 16-wide Dh slab ----
        if (wid < 4) {
            const _Float16* vb = v_t + (bh * DH + (size_t)wid * 16) * NSEQ;
            v8f acc = {};
            for (int k0 = 0; k0 < NSEQ; k0 += 32) {
                v16h a = load_a_frag_lds(sc, NSEQ, k0, lane);
                v16h bm = load_b_frag(vb, NSEQ, k0, lane);
                acc = wmma_f16(a, bm, acc);
            }
            int n = lane & 15, g = lane >> 4;
#pragma unroll
            for (int i = 0; i < 8; ++i) {
                int nq = qt * 16 + g * 8 + i;
                ctx_h[((size_t)(b * NSEQ + nq) * NHEAD + h) * DH + wid * 16 + n] =
                    (_Float16)acc[i];
            }
        }
        __syncthreads();   // sc reused next batch iteration
    }

    // ---- score_delta = sum_b mask, written once ----
    for (int idx = tid; idx < 16 * 1024; idx += 256) {
        int r = idx >> 10, kk = idx & 1023;
        score_delta[((size_t)h * NSEQ + qt * 16 + r) * NSEQ + kk] =
            (float)cnt[idx];
    }
}

// ---------------------------------------------------------------------------
// Kernel 3: output projection. [4096 x 768] @ [768 x 768]^T + bias -> f32 out
// ---------------------------------------------------------------------------
__global__ void ucb_proj_gemm_kernel(const _Float16* __restrict__ ctx_h,
                                     const _Float16* __restrict__ pw_h,
                                     const float* __restrict__ pb,
                                     float* __restrict__ out) {
    int lane = threadIdx.x & 31, wid = threadIdx.x >> 5;
    int ntile = blockIdx.x * 8 + wid;   // 0..47
    int mtile = blockIdx.y;             // 0..255
    const _Float16* abase = ctx_h + (size_t)mtile * 16 * CDIM;
    const _Float16* bbase = pw_h + (size_t)ntile * 16 * CDIM;
    v8f acc = {};
    for (int k0 = 0; k0 < CDIM; k0 += 32) {
        v16h a = load_a_frag(abase, CDIM, k0, lane);
        v16h b = load_b_frag(bbase, CDIM, k0, lane);
        acc = wmma_f16(a, b, acc);
    }
    int n = lane & 15, g = lane >> 4;
    int col = ntile * 16 + n;
    float bias = pb[col];
#pragma unroll
    for (int i = 0; i < 8; ++i) {
        int row = mtile * 16 + g * 8 + i;
        out[(size_t)row * CDIM + col] = acc[i] + bias;
    }
}

// ---------------------------------------------------------------------------
// Launch
// ---------------------------------------------------------------------------
extern "C" void kernel_launch(void* const* d_in, const int* in_sizes, int n_in,
                              void* d_out, int out_size, void* d_ws, size_t ws_size,
                              hipStream_t stream) {
    const float* x      = (const float*)d_in[0];
    const float* qkv_w  = (const float*)d_in[1];
    const float* proj_w = (const float*)d_in[2];
    const float* proj_b = (const float*)d_in[3];
    const float* ucb    = (const float*)d_in[4];
    const int* counter  = (const int*)d_in[5];
    const int* enabled  = (const int*)d_in[6];

    float* out = (float*)d_out;
    float* score_delta = out + (size_t)BATCH * NSEQ * CDIM;

    // workspace carve-up (f16 halves)
    const size_t XH  = (size_t)BATCH * NSEQ * CDIM;   // 3,145,728
    const size_t WH  = (size_t)QKVDIM * CDIM;         // 1,769,472
    const size_t PWH = (size_t)CDIM * CDIM;           //   589,824
    const size_t QKV = (size_t)BATCH * NHEAD * NSEQ * DH; // 3,145,728
    _Float16* ws    = (_Float16*)d_ws;
    _Float16* x_h   = ws;
    _Float16* w_h   = x_h + XH;
    _Float16* pw_h  = w_h + WH;
    _Float16* q_h   = pw_h + PWH;
    _Float16* k_h   = q_h + QKV;
    _Float16* v_t   = k_h + QKV;
    _Float16* ctx_h = v_t + QKV;

    // 0) f32 -> f16 staging
    ucb_cvt_f16_kernel<<<2048, 256, 0, stream>>>(x, x_h, (int)XH);
    ucb_cvt_f16_kernel<<<2048, 256, 0, stream>>>(qkv_w, w_h, (int)WH);
    ucb_cvt_f16_kernel<<<1024, 256, 0, stream>>>(proj_w, pw_h, (int)PWH);

    // 1) QKV projection
    ucb_qkv_gemm_kernel<<<dim3(18, 256), 256, 0, stream>>>(x_h, w_h, q_h, k_h, v_t);

    // 2) fused attention + UCB top-k + score_delta
    size_t smem = (size_t)16 * 1024 * 4 * 2 + 16 * 1024;  // 147456 B (<320KB WGP LDS)
    hipFuncSetAttribute(reinterpret_cast<const void*>(ucb_attn_kernel),
                        hipFuncAttributeMaxDynamicSharedMemorySize, (int)smem);
    ucb_attn_kernel<<<dim3(64, 12), 256, smem, stream>>>(
        q_h, k_h, v_t, ucb, counter, enabled, ctx_h, score_delta);

    // 3) output projection
    ucb_proj_gemm_kernel<<<dim3(6, 256), 256, 0, stream>>>(ctx_h, pw_h, proj_b, out);
}
